// ArcFace_52939766890862
// MI455X (gfx1250) — matrix operand order repaired and checked by reading, not compiled, not run on hardware
//
#include <hip/hip_runtime.h>
#include <math.h>

// ---------------- problem constants (from reference) ----------------
constexpr int   Bn      = 1024;     // batch rows
constexpr int   Dk      = 512;      // embedding dim (K)
constexpr int   NC      = 50000;    // classes (N)
constexpr float S_SCALE = 64.0f;
constexpr float EPS     = 1e-7f;
constexpr float COS_M   = 0.8775825618903728f;  // cos(0.5)
constexpr float SIN_M   = 0.479425538604203f;   // sin(0.5)

typedef __attribute__((ext_vector_type(16))) __bf16 v16bf;
typedef __attribute__((ext_vector_type(8)))  float  v8f;

__device__ __forceinline__ unsigned short f32_to_bf16(float f) {
    union { float f; unsigned int u; } c; c.f = f;
    unsigned int u = c.u;
    // round-to-nearest-even
    unsigned int r = (u + 0x7FFFu + ((u >> 16) & 1u)) >> 16;
    return (unsigned short)r;
}

// ---------------- 1) column inverse norms of W ----------------
__global__ __launch_bounds__(256)
void colnorm_kernel(const float* __restrict__ W, float* __restrict__ wInv) {
    int n = blockIdx.x * 256 + threadIdx.x;
    if (n >= NC) return;
    float s = 0.f;
    #pragma unroll 8
    for (int k = 0; k < Dk; ++k) {
        float v = W[(size_t)k * NC + n];   // coalesced across threads
        s = fmaf(v, v, s);
    }
    wInv[n] = rsqrtf(s);
}

// ---------------- 2) row-normalize x -> bf16 ----------------
__global__ __launch_bounds__(256)
void xnorm_kernel(const float* __restrict__ x, unsigned short* __restrict__ xn) {
    int row = blockIdx.x, tid = threadIdx.x;
    float v0 = x[(size_t)row * Dk + tid];
    float v1 = x[(size_t)row * Dk + 256 + tid];
    __shared__ float red[256];
    red[tid] = v0 * v0 + v1 * v1;
    __syncthreads();
    for (int off = 128; off > 0; off >>= 1) {
        if (tid < off) red[tid] += red[tid + off];
        __syncthreads();
    }
    float inv = rsqrtf(red[0]);
    xn[(size_t)row * Dk + tid]       = f32_to_bf16(v0 * inv);
    xn[(size_t)row * Dk + 256 + tid] = f32_to_bf16(v1 * inv);
}

// ---------------- 3) normalize + transpose W -> WnT[NC][Dk] bf16 ----------------
__global__ __launch_bounds__(256)
void wtrans_kernel(const float* __restrict__ W, const float* __restrict__ wInv,
                   unsigned short* __restrict__ wnt) {
    __shared__ unsigned short tile[64][65];          // [k][n], padded
    const int n0 = blockIdx.x * 64;
    const int k0 = blockIdx.y * 64;
    const int c  = threadIdx.x & 63;                 // n offset within tile
    const int r0 = threadIdx.x >> 6;                 // 0..3
    const int n  = n0 + c;
    const float inv = (n < NC) ? wInv[n] : 0.f;
    #pragma unroll
    for (int i = 0; i < 16; ++i) {
        int r = r0 + i * 4;                          // k offset
        float v = (n < NC) ? W[(size_t)(k0 + r) * NC + n] : 0.f;  // coalesced (L2-hot)
        tile[r][c] = f32_to_bf16(v * inv);
    }
    __syncthreads();
    const int rr = threadIdx.x >> 2;                 // output n row 0..63
    const int cc = (threadIdx.x & 3) * 16;           // k chunk
    const int nn = n0 + rr;
    if (nn < NC) {
        unsigned short* dst = wnt + (size_t)nn * Dk + k0 + cc;
        #pragma unroll
        for (int j = 0; j < 16; ++j) dst[j] = tile[cc + j][rr];   // 32B/thread, coalesced
    }
}

// ---------------- 4) extract labels from one-hot y, zero row sums ----------------
__global__ __launch_bounds__(256)
void label_kernel(const float* __restrict__ y, int* __restrict__ label,
                  float* __restrict__ rowSum) {
    int row = blockIdx.x;
    if (threadIdx.x == 0) rowSum[row] = 0.f;
    const float* yr = y + (size_t)row * NC;
    for (int i = threadIdx.x; i < NC; i += 256)
        if (yr[i] > 0.5f) label[row] = i;            // exactly one hit per row
}

// ---------------- 5) fused WMMA GEMM + ArcFace margin + exp + row partial sums ----
// Block = 256 threads = 8 waves; wave w -> rows [m0, m0+32), block cols [n0, n0+64).
// Wave tile 32x64: 2 A-frags x 4 B-frags -> 8 WMMA per K-step from 6KB of loads
// (0.75 KB operand traffic per WMMA vs 1.25 KB for a 16x64 tile).
__global__ __launch_bounds__(256)
void gemm_arcface_kernel(const unsigned short* __restrict__ xn,
                         const unsigned short* __restrict__ wnt,
                         const int* __restrict__ label,
                         float* __restrict__ rowSum,
                         float* __restrict__ out) {
    const int tid  = threadIdx.x;
    const int wave = tid >> 5;
    const int lane = tid & 31;
    const int l15  = lane & 15;
    const int kh   = lane >> 4;                      // which 16-lane half
    const int m0   = blockIdx.y * 256 + wave * 32;   // 32 rows per wave
    const int n0   = blockIdx.x * 64;

    union Frag { v16bf v; uint4 q[2]; };
    Frag a[2];
    Frag b[4];
    v8f acc[2][4] = {};                              // [A-group][N-tile], f32, zeroed

    // A: row-major xn; lane needs K = kh*8 + [0,8) and 16 + kh*8 + [0,8)  (ISA 7.12.2)
    const char* aBase0 = (const char*)(xn + (size_t)(m0 + l15) * Dk);
    const char* aBase1 = (const char*)(xn + (size_t)(m0 + 16 + l15) * Dk);
    // B: WnT row-major over K; lane needs K = kh*8 + [0,8) and 16 + kh*8 + [0,8)
    const char* bBase[4];
    #pragma unroll
    for (int t = 0; t < 4; ++t) {
        int n = n0 + t * 16 + l15;
        if (n >= NC) n = NC - 1;                     // clamp; masked at epilogue
        bBase[t] = (const char*)(wnt + (size_t)n * Dk);
    }

    for (int kk = 0; kk < Dk; kk += 32) {
        a[0].q[0] = *(const uint4*)(aBase0 + 2 * (kk + kh * 8));
        a[0].q[1] = *(const uint4*)(aBase0 + 2 * (kk + 16 + kh * 8));
        a[1].q[0] = *(const uint4*)(aBase1 + 2 * (kk + kh * 8));
        a[1].q[1] = *(const uint4*)(aBase1 + 2 * (kk + 16 + kh * 8));
        #pragma unroll
        for (int t = 0; t < 4; ++t) {
            const char* p = bBase[t] + 2 * (kk + kh * 8);
            b[t].q[0] = *(const uint4*)(p);
            b[t].q[1] = *(const uint4*)(p + 32);
        }
        #pragma unroll
        for (int g = 0; g < 2; ++g)
            #pragma unroll
            for (int t = 0; t < 4; ++t)
                acc[g][t] = __builtin_amdgcn_wmma_f32_16x16x32_bf16(
                                false, a[g].v, false, b[t].v, (short)0, acc[g][t],
                                false, false);
    }

    // Epilogue. C/D layout: VGPR r -> M = m0 + g*16 + kh*8 + r ; N = n0 + t*16 + l15.
    #pragma unroll
    for (int g = 0; g < 2; ++g) {
        #pragma unroll
        for (int r = 0; r < 8; ++r) {
            const int m   = m0 + g * 16 + kh * 8 + r;
            const int lab = label[m];
            float s = 0.f;
            #pragma unroll
            for (int t = 0; t < 4; ++t) {
                const int n = n0 + t * 16 + l15;
                float c = acc[g][t][r];
                if (n == lab) {
                    float cc = fminf(fmaxf(c, -1.f + EPS), 1.f - EPS);
                    c = cc * COS_M - sqrtf(fmaxf(1.f - cc * cc, 0.f)) * SIN_M;
                }
                float e = __expf(S_SCALE * c);       // safe: |64*c| <= ~65 << f32 range
                if (n < NC) {
                    out[(size_t)m * NC + n] = e;
                    s += e;
                }
            }
            // reduce across the 16 lanes that share row m
            #pragma unroll
            for (int off = 1; off < 16; off <<= 1) s += __shfl_xor(s, off, 32);
            if (l15 == 0) atomicAdd(&rowSum[m], s);
        }
    }
}

// ---------------- 6) divide by row sums ----------------
__global__ __launch_bounds__(256)
void scale_kernel(float* __restrict__ out, const float* __restrict__ rowSum) {
    const int row = blockIdx.y;
    const int idx = blockIdx.x * 256 + threadIdx.x;
    constexpr int NQ = NC / 4;                       // 12500 float4 per row
    if (idx < NQ) {
        float inv = 1.f / rowSum[row];
        float4* p = (float4*)out + (size_t)row * NQ + idx;
        float4 v = *p;
        v.x *= inv; v.y *= inv; v.z *= inv; v.w *= inv;
        *p = v;
    }
}

// ---------------- launcher ----------------
extern "C" void kernel_launch(void* const* d_in, const int* in_sizes, int n_in,
                              void* d_out, int out_size, void* d_ws, size_t ws_size,
                              hipStream_t stream) {
    (void)in_sizes; (void)n_in; (void)out_size; (void)ws_size;
    const float* x = (const float*)d_in[0];          // [Bn, Dk]
    const float* y = (const float*)d_in[1];          // [Bn, NC] one-hot
    const float* W = (const float*)d_in[2];          // [Dk, NC]
    float* out = (float*)d_out;                      // [Bn, NC]

    // workspace carve-out (256B aligned)
    char* ws = (char*)d_ws;
    size_t off = 0;
    auto carve = [&](size_t bytes) { char* p = ws + off; off = (off + bytes + 255) & ~(size_t)255; return p; };
    unsigned short* xn   = (unsigned short*)carve((size_t)Bn * Dk * 2);   // 1 MB
    unsigned short* wnt  = (unsigned short*)carve((size_t)NC * Dk * 2);   // 51.2 MB
    float*          wInv = (float*)carve((size_t)NC * 4);
    int*            lab  = (int*)carve((size_t)Bn * 4);
    float*          rsum = (float*)carve((size_t)Bn * 4);

    const int nTiles64 = (NC + 63) / 64;             // 782

    colnorm_kernel<<<dim3((NC + 255) / 256), dim3(256), 0, stream>>>(W, wInv);
    xnorm_kernel  <<<dim3(Bn),               dim3(256), 0, stream>>>(x, xn);
    wtrans_kernel <<<dim3(nTiles64, Dk / 64), dim3(256), 0, stream>>>(W, wInv, wnt);
    label_kernel  <<<dim3(Bn),               dim3(256), 0, stream>>>(y, lab, rsum);
    gemm_arcface_kernel<<<dim3(nTiles64, Bn / 256), dim3(256), 0, stream>>>(xn, wnt, lab, rsum, out);
    scale_kernel  <<<dim3((NC / 4 + 255) / 256, Bn), dim3(256), 0, stream>>>(out, rsum);
}
